// BitRowParallelLinear_18580028523111
// MI455X (gfx1250) — matrix operand order repaired and checked by reading, not compiled, not run on hardware
//
#include <hip/hip_runtime.h>
#include <hip/hip_bf16.h>

// ---------------------------------------------------------------------------
// BitNet-style linear: out[M,N] = x[M,K] @ sign(W)[K,N] + bias[N]
//   M = B*S = 8192, N = D_OUT = 4096, K = D_IN = 4096
// Phase 0: x f32 -> bf16 into d_ws[0, 64MB)
// Phase 1: W f32 -> sign(W) bf16 into d_ws[64MB, 96MB)
// Phase 2: double-buffered bf16 WMMA GEMM, 64x64 tile per wave,
//          16 v_wmma_f32_16x16x32_bf16 per K-step. Tile staging uses
//          GLOBAL_LOAD_ASYNC_TO_LDS_B128 (ASYNCcnt) when the builtin exists,
//          else falls back to register staging.
// ---------------------------------------------------------------------------

typedef __attribute__((ext_vector_type(16))) __bf16 v16bf;
typedef __attribute__((ext_vector_type(8)))  float  v8f;
typedef __attribute__((__vector_size__(16))) int    vi4;   // builtin's int4

#if __has_builtin(__builtin_amdgcn_global_load_async_to_lds_b128)
#define USE_ASYNC_LDS 1
#else
#define USE_ASYNC_LDS 0
#endif

#define GLOBAL_AS __attribute__((address_space(1)))
#define LDS_AS    __attribute__((address_space(3)))

namespace {
constexpr int M_DIM = 8192;   // B * S
constexpr int N_DIM = 4096;   // D_OUT
constexpr int K_DIM = 4096;   // D_IN

constexpr int BM = 256;       // block tile M
constexpr int BN = 128;       // block tile N
constexpr int BK = 32;        // K depth per LDS stage
constexpr int LDK = BK + 8;   // 40 ushorts -> 80B row stride (16B aligned)
}

// round-to-nearest-even f32 -> bf16 bits
__device__ __forceinline__ unsigned short f32_to_bf16_bits(float f) {
    unsigned int u = __builtin_bit_cast(unsigned int, f);
    unsigned int rounding = 0x7FFFu + ((u >> 16) & 1u);
    return (unsigned short)((u + rounding) >> 16);
}

#if USE_ASYNC_LDS
// 16B per lane, global -> LDS via the async engine (ASYNCcnt)
__device__ __forceinline__ void async_copy_b128(const unsigned short* g,
                                                unsigned short* l) {
    unsigned short* gm = const_cast<unsigned short*>(g);
    __builtin_amdgcn_global_load_async_to_lds_b128(
        (GLOBAL_AS vi4*)gm, (LDS_AS vi4*)l, /*offset=*/0, /*cpol=*/0);
}

__device__ __forceinline__ void wait_async_zero() {
#if __has_builtin(__builtin_amdgcn_s_wait_asynccnt)
    __builtin_amdgcn_s_wait_asynccnt(0);
#else
    asm volatile("s_wait_asynccnt 0x0" ::: "memory");
#endif
}
#endif

// ------------------------- Phase 0: x f32 -> bf16 --------------------------
__global__ __launch_bounds__(256)
void bitnet_x_to_bf16_kernel(const float* __restrict__ x,
                             unsigned short* __restrict__ xb,
                             int n4) {
    int i = blockIdx.x * blockDim.x + threadIdx.x;
    if (i >= n4) return;
    float4 v = reinterpret_cast<const float4*>(x)[i];
    uint2 packed;
    packed.x = (unsigned int)f32_to_bf16_bits(v.x) |
               ((unsigned int)f32_to_bf16_bits(v.y) << 16);
    packed.y = (unsigned int)f32_to_bf16_bits(v.z) |
               ((unsigned int)f32_to_bf16_bits(v.w) << 16);
    reinterpret_cast<uint2*>(xb)[i] = packed;
}

// ------------------------- Phase 1: W -> sign(W) bf16 ----------------------
__global__ __launch_bounds__(256)
void bitnet_sign_quant_kernel(const float* __restrict__ w,
                              unsigned short* __restrict__ wsign,
                              int n4) {
    int i = blockIdx.x * blockDim.x + threadIdx.x;
    if (i >= n4) return;
    float4 v = reinterpret_cast<const float4*>(w)[i];
    // sign() in bf16: +1 = 0x3F80, -1 = 0xBF80, 0 = 0x0000 (exact)
    unsigned int s0 = (v.x > 0.f) ? 0x3F80u : (v.x < 0.f ? 0xBF80u : 0u);
    unsigned int s1 = (v.y > 0.f) ? 0x3F80u : (v.y < 0.f ? 0xBF80u : 0u);
    unsigned int s2 = (v.z > 0.f) ? 0x3F80u : (v.z < 0.f ? 0xBF80u : 0u);
    unsigned int s3 = (v.w > 0.f) ? 0x3F80u : (v.w < 0.f ? 0xBF80u : 0u);
    uint2 packed;
    packed.x = s0 | (s1 << 16);
    packed.y = s2 | (s3 << 16);
    reinterpret_cast<uint2*>(wsign)[i] = packed;
}

// ------------------------- Phase 2: WMMA GEMM ------------------------------
// Grid: (M/BM, N/BN) = (32, 32). Block: 256 threads = 8 wave32 in a 4x2 grid.
// Each wave computes a 64x64 sub-tile = 4x4 tiles of 16x16 (v8f accum each).
__global__ __launch_bounds__(256)
void bitnet_wmma_gemm_kernel(const unsigned short* __restrict__ xb,    // bf16 bits, [M][K]
                             const unsigned short* __restrict__ wsign, // bf16 bits, [N][K]
                             const float* __restrict__ bias,
                             float* __restrict__ out) {
    __shared__ unsigned short As[2][BM * LDK];   // x tile,     [BM][BK] bf16
    __shared__ unsigned short Bs[2][BN * LDK];   // signW tile, [BN][BK] bf16

    const int tid  = threadIdx.x;
    const int lane = tid & 31;
    const int wave = tid >> 5;
    const int waveM = wave >> 1;          // 0..3 -> 64-row band
    const int waveN = wave & 1;           // 0..1 -> 64-col band

    const int blockRow = blockIdx.x * BM; // M offset
    const int blockCol = blockIdx.y * BN; // N offset

    v8f acc[4][4];
#pragma unroll
    for (int i = 0; i < 4; ++i)
#pragma unroll
        for (int j = 0; j < 4; ++j)
            acc[i][j] = v8f{};

    // --- cooperative loader mapping ---
    // A tile: 256 rows x 32 cols bf16; thread t owns row t (4 x b128)
    const unsigned short* aSrc = xb + (size_t)(blockRow + tid) * K_DIM;
    // B tile: 128 rows x 32 cols bf16; thread t owns half of row t>>1 (2 x b128)
    const int bRow = tid >> 1;
    const int bCol = (tid & 1) * 16;
    const unsigned short* bSrc = wsign + (size_t)(blockCol + bRow) * K_DIM + bCol;

    unsigned short* aDst = &As[0][tid * LDK];
    unsigned short* bDst = &Bs[0][bRow * LDK + bCol];
    const int bufStrideA = BM * LDK;
    const int bufStrideB = BN * LDK;

#if USE_ASYNC_LDS
    // stage a K-tile into LDS buffer `b` via the async engine
    auto stage_tile = [&](int k0, int b) {
        unsigned short* ad = aDst + b * bufStrideA;
        async_copy_b128(aSrc + k0,      ad);
        async_copy_b128(aSrc + k0 + 8,  ad + 8);
        async_copy_b128(aSrc + k0 + 16, ad + 16);
        async_copy_b128(aSrc + k0 + 24, ad + 24);
        unsigned short* bd = bDst + b * bufStrideB;
        async_copy_b128(bSrc + k0,     bd);
        async_copy_b128(bSrc + k0 + 8, bd + 8);
    };

    // --- prologue: tile 0 -> buffer 0 ---
    stage_tile(0, 0);
    wait_async_zero();
    __syncthreads();
#else
    // --- prologue (fallback): register staging ---
    {
        const uint4* ap = reinterpret_cast<const uint4*>(aSrc);
        uint4 a0 = ap[0], a1 = ap[1], a2 = ap[2], a3 = ap[3];
        const uint4* bp = reinterpret_cast<const uint4*>(bSrc);
        uint4 b0 = bp[0], b1 = bp[1];
        uint4* ad = reinterpret_cast<uint4*>(aDst);
        ad[0] = a0; ad[1] = a1; ad[2] = a2; ad[3] = a3;
        uint4* bd = reinterpret_cast<uint4*>(bDst);
        bd[0] = b0; bd[1] = b1;
    }
    __syncthreads();
#endif

    int buf = 0;
    for (int k0 = 0; k0 < K_DIM; k0 += BK) {
        const bool hasNext = (k0 + BK) < K_DIM;

#if USE_ASYNC_LDS
        // ---- kick off async staging of tile k+1 (overlaps the WMMAs) ----
        if (hasNext)
            stage_tile(k0 + BK, buf ^ 1);
#else
        uint4 a0, a1, a2, a3, b0, b1;
        if (hasNext) {
            const uint4* ap = reinterpret_cast<const uint4*>(aSrc + k0 + BK);
            a0 = ap[0]; a1 = ap[1]; a2 = ap[2]; a3 = ap[3];
            const uint4* bp = reinterpret_cast<const uint4*>(bSrc + k0 + BK);
            b0 = bp[0]; b1 = bp[1];
        }
#endif

        // ---- fragment loads per CDNA5 wave32 WMMA VGPR layouts ----
        const unsigned short* aBase = &As[0][buf * bufStrideA];
        const unsigned short* bBase = &Bs[0][buf * bufStrideB];
        const int r   = lane & 15;
        const int kbA = (lane >> 4) * 8;   // A: lanes 0-15 K={0..7,16..23}; 16-31 K={8..15,24..31}
        const int kbB = (lane >> 4) * 8;   // B: lanes 0-15 K={0..7,16..23}; 16-31 K={8..15,24..31}

        v16bf afrag[4];
#pragma unroll
        for (int i = 0; i < 4; ++i) {
            const unsigned short* p = aBase + (waveM * 64 + i * 16 + r) * LDK;
            union { uint4 u[2]; v16bf v; } f;
            f.u[0] = *reinterpret_cast<const uint4*>(p + kbA);
            f.u[1] = *reinterpret_cast<const uint4*>(p + kbA + 16);
            afrag[i] = f.v;
        }
        v16bf bfrag[4];
#pragma unroll
        for (int j = 0; j < 4; ++j) {
            const unsigned short* p = bBase + (waveN * 64 + j * 16 + r) * LDK;
            union { uint4 u[2]; v16bf v; } f;
            f.u[0] = *reinterpret_cast<const uint4*>(p + kbB);
            f.u[1] = *reinterpret_cast<const uint4*>(p + kbB + 16);
            bfrag[j] = f.v;
        }

        // ---- 16 WMMAs: D = A x B + C ----
#pragma unroll
        for (int i = 0; i < 4; ++i)
#pragma unroll
            for (int j = 0; j < 4; ++j)
                acc[i][j] = __builtin_amdgcn_wmma_f32_16x16x32_bf16(
                    /*neg_a=*/false, afrag[i],
                    /*neg_b=*/false, bfrag[j],
                    /*c_mod=*/(short)0, acc[i][j],
                    /*reuse_a=*/false, /*reuse_b=*/false);

#if USE_ASYNC_LDS
        // own async transfers complete, then workgroup-wide visibility
        if (hasNext)
            wait_async_zero();
#else
        if (hasNext) {
            uint4* ad = reinterpret_cast<uint4*>(aDst + (buf ^ 1) * bufStrideA);
            ad[0] = a0; ad[1] = a1; ad[2] = a2; ad[3] = a3;
            uint4* bd = reinterpret_cast<uint4*>(bDst + (buf ^ 1) * bufStrideB);
            bd[0] = b0; bd[1] = b1;
        }
#endif
        __syncthreads();
        buf ^= 1;
    }

    // ---- epilogue: C/D layout -> global + bias ----
    // VGPR r: lanes 0-15 -> M = r, N = lane; lanes 16-31 -> M = r+8, N = lane-16
    const int rbase = (lane >> 4) * 8;
    const int colIn = lane & 15;
#pragma unroll
    for (int i = 0; i < 4; ++i) {
#pragma unroll
        for (int j = 0; j < 4; ++j) {
            const int gcol = blockCol + waveN * 64 + j * 16 + colIn;
            const float bv = bias[gcol];
#pragma unroll
            for (int rr = 0; rr < 8; ++rr) {
                const int grow = blockRow + waveM * 64 + i * 16 + rbase + rr;
                out[(size_t)grow * N_DIM + gcol] = acc[i][j][rr] + bv;
            }
        }
    }
}

// ---------------------------------------------------------------------------
extern "C" void kernel_launch(void* const* d_in, const int* in_sizes, int n_in,
                              void* d_out, int out_size, void* d_ws, size_t ws_size,
                              hipStream_t stream) {
    const float* x    = (const float*)d_in[0];   // [8192, 4096] f32 (B,S flattened)
    const float* w    = (const float*)d_in[1];   // [4096, 4096] f32
    const float* bias = (const float*)d_in[2];   // [4096] f32
    float* out = (float*)d_out;                  // [8192, 4096] f32

    // d_ws layout: [0, 64MB) x_bf16 ; [64MB, 96MB) sign(W) bf16
    unsigned short* xb    = (unsigned short*)d_ws;
    unsigned short* wsign = xb + (size_t)M_DIM * K_DIM;

    // Phase 0: x -> bf16 (32M elems, float4 granularity)
    const int nx4 = (M_DIM * K_DIM) / 4;
    bitnet_x_to_bf16_kernel<<<nx4 / 256, 256, 0, stream>>>(x, xb, nx4);

    // Phase 1: W -> +/-1 bf16 (16M elems)
    const int nw4 = (N_DIM * K_DIM) / 4;
    bitnet_sign_quant_kernel<<<nw4 / 256, 256, 0, stream>>>(w, wsign, nw4);

    // Phase 2: WMMA GEMM
    dim3 grid(M_DIM / BM, N_DIM / BN);           // (32, 32)
    bitnet_wmma_gemm_kernel<<<grid, 256, 0, stream>>>(xb, wsign, bias, out);
}
